// NeuronGemma2DecoderLayer_42726334660784
// MI455X (gfx1250) — matrix operand order repaired and checked by reading, not compiled, not run on hardware
//
#include <hip/hip_runtime.h>
#include <hip/hip_bf16.h>

// ---------------------------------------------------------------------------
// Gemma2 decoder layer for gfx1250 (MI455X).  bf16 WMMA everywhere.
// GEMM: 128x128 block tile, 8 waves x (32x64), double-buffered LDS staging.
// Attention: flash-style streaming with softcap + sliding-window mask.
// ---------------------------------------------------------------------------

typedef __attribute__((ext_vector_type(16))) __bf16 v16bf;
typedef __attribute__((ext_vector_type(8)))  __bf16 v8bf;
typedef __attribute__((ext_vector_type(8)))  float  v8f;
typedef __attribute__((ext_vector_type(4)))  float  f32x4;

static constexpr int cS   = 2048;
static constexpr int cH   = 3584;
static constexpr int cNH  = 16;
static constexpr int cNKV = 8;
static constexpr int cHD  = 256;
static constexpr int cI   = 14336;
static constexpr int cDQ  = cNH * cHD;   // 4096
static constexpr int cDKV = cNKV * cHD;  // 2048
static constexpr float cSCALE   = 0.0625f;        // 256^-0.5
static constexpr float cSOFTCAP = 50.0f;
static constexpr int   cWINDOW  = 4096;
static constexpr float cNEG     = -3.0e38f;

union FragU { v16bf v; struct { v8bf lo, hi; } p; };
union PackU { unsigned u; __bf16 b[2]; };

// ---------------------------------------------------------------------------
// block reduction helper (256 threads = 8 waves of 32)
// ---------------------------------------------------------------------------
__device__ inline float block_reduce_sum_256(float v, float* red) {
  #pragma unroll
  for (int m = 16; m; m >>= 1) v += __shfl_xor(v, m, 32);
  if ((threadIdx.x & 31) == 0) red[threadIdx.x >> 5] = v;
  __syncthreads();
  float tot = 0.f;
  #pragma unroll
  for (int i = 0; i < 8; ++i) tot += red[i];
  return tot;
}

// ---------------------------------------------------------------------------
// RMSNorm -> bf16 (GEMM activation input)
// ---------------------------------------------------------------------------
__global__ __launch_bounds__(256) void rmsnorm_bf16_kernel(
    const float* __restrict__ x, const float* __restrict__ w,
    __bf16* __restrict__ out, int Hn) {
  __shared__ float red[8];
  int row = blockIdx.x;
  const float* xr = x + (size_t)row * Hn;
  float ss = 0.f;
  for (int i = threadIdx.x; i < Hn; i += 256) { float v = xr[i]; ss += v * v; }
  float tot = block_reduce_sum_256(ss, red);
  float rstd = __frsqrt_rn(tot / (float)Hn + 1e-6f);
  __bf16* orow = out + (size_t)row * Hn;
  for (int i = threadIdx.x; i < Hn; i += 256)
    orow[i] = (__bf16)(xr[i] * rstd * w[i]);
}

// ---------------------------------------------------------------------------
// out = x + rmsnorm(y, w)   (fp32 out; used post-attn and as final output)
// ---------------------------------------------------------------------------
__global__ __launch_bounds__(256) void add_rmsnorm_kernel(
    const float* __restrict__ x, const float* __restrict__ y,
    const float* __restrict__ w, float* __restrict__ out, int Hn) {
  __shared__ float red[8];
  int row = blockIdx.x;
  const float* xr = x + (size_t)row * Hn;
  const float* yr = y + (size_t)row * Hn;
  float ss = 0.f;
  for (int i = threadIdx.x; i < Hn; i += 256) { float v = yr[i]; ss += v * v; }
  float tot = block_reduce_sum_256(ss, red);
  float rstd = __frsqrt_rn(tot / (float)Hn + 1e-6f);
  float* orow = out + (size_t)row * Hn;
  for (int i = threadIdx.x; i < Hn; i += 256)
    orow[i] = xr[i] + yr[i] * rstd * w[i];
}

// ---------------------------------------------------------------------------
// WMMA GEMM: C[M,N] f32 = A[M,K] bf16 (row-major) x B[K,N] f32 weights
// (converted to bf16 while staging; Bs kept K-contiguous for B-fragments).
// Block tile 128x128, 8 waves in 4(M)x2(N), each wave 32x64 (2x4 WMMA tiles).
// Double-buffered LDS: next K-tile fetched to registers during WMMA, committed
// after compute; one barrier per stage.  M%128==N%128==0, K%32==0 here.
// ---------------------------------------------------------------------------
__global__ __launch_bounds__(256) void wmma_gemm_kernel(
    const __bf16* __restrict__ A, const float* __restrict__ B,
    float* __restrict__ C, int M, int N, int K) {
  __shared__ __align__(32) __bf16 As[2][128][32];   // 16 KB
  __shared__ __align__(32) __bf16 Bs[2][128][32];   // 16 KB, Bs[n][k]
  const int tid  = threadIdx.x;
  const int bm   = blockIdx.y * 128;
  const int bn   = blockIdx.x * 128;
  const int wv   = tid >> 5;
  const int lane = tid & 31;
  const int ln   = lane & 15;
  const int wm   = (wv & 3) * 32;       // 4 waves along M
  const int wn   = (wv >> 2) * 64;      // 2 waves along N
  const int kbA  = (lane >= 16) ? 16 : 0;
  const int kbB  = (lane >= 16) ? 16 : 0;

  // staging assignments
  const int arow = tid >> 1, ahk = (tid & 1) * 16;   // A: 32B per thread
  const int bkr  = (tid >> 4) * 2;                   // B: k-row pair (even)
  const int bn0  = (tid & 15) * 8;                   // B: 8 columns

  v8f acc[2][4] = {};

  // ---- prologue: stage K-tile 0 into buffer 0 ----
  {
    const __bf16* srcA = A + (size_t)(bm + arow) * K + ahk;
    *(f32x4*)&As[0][arow][ahk]     = *(const f32x4*)(srcA);
    *(f32x4*)&As[0][arow][ahk + 8] = *(const f32x4*)(srcA + 8);
    const float* srcB0 = B + (size_t)bkr * N + bn + bn0;
    #pragma unroll
    for (int j = 0; j < 8; ++j) {
      PackU pk;
      pk.b[0] = (__bf16)srcB0[j];
      pk.b[1] = (__bf16)srcB0[N + j];
      *(unsigned*)&Bs[0][bn0 + j][bkr] = pk.u;
    }
  }
  __syncthreads();

  int p = 0;
  for (int k0 = 0; k0 < K; k0 += 32) {
    const int nxt = p ^ 1;
    const bool has_next = (k0 + 32) < K;

    // --- fetch next K-tile into registers (overlaps with WMMA below) ---
    f32x4 ra0, ra1, rb0, rb1, rb2, rb3;
    if (has_next) {
      const __bf16* srcA = A + (size_t)(bm + arow) * K + (k0 + 32) + ahk;
      ra0 = *(const f32x4*)(srcA);
      ra1 = *(const f32x4*)(srcA + 8);
      const float* srcB0 = B + (size_t)(k0 + 32 + bkr) * N + bn + bn0;
      rb0 = *(const f32x4*)(srcB0);
      rb1 = *(const f32x4*)(srcB0 + 4);
      rb2 = *(const f32x4*)(srcB0 + N);
      rb3 = *(const f32x4*)(srcB0 + N + 4);
      if (k0 + 64 < K) {
        __builtin_prefetch(A + (size_t)(bm + arow) * K + k0 + 64, 0, 1);
        __builtin_prefetch(B + (size_t)(k0 + 64 + bkr) * N + bn, 0, 1);
      }
    }

    // --- load fragments once, then 8 WMMAs ---
    FragU a[2];
    #pragma unroll
    for (int mi = 0; mi < 2; ++mi) {
      const int m = wm + mi * 16 + ln;
      a[mi].p.lo = *(const v8bf*)&As[p][m][kbA];
      a[mi].p.hi = *(const v8bf*)&As[p][m][kbA + 8];
    }
    v16bf b[4];
    #pragma unroll
    for (int ni = 0; ni < 4; ++ni)
      b[ni] = *(const v16bf*)&Bs[p][wn + ni * 16 + ln][kbB];
    #pragma unroll
    for (int mi = 0; mi < 2; ++mi)
      #pragma unroll
      for (int ni = 0; ni < 4; ++ni)
        acc[mi][ni] = __builtin_amdgcn_wmma_f32_16x16x32_bf16(
            false, a[mi].v, false, b[ni], (short)0, acc[mi][ni], false, false);

    // --- commit next tile to the alternate buffer ---
    if (has_next) {
      *(f32x4*)&As[nxt][arow][ahk]     = ra0;
      *(f32x4*)&As[nxt][arow][ahk + 8] = ra1;
      #pragma unroll
      for (int j = 0; j < 4; ++j) {
        PackU pk;
        pk.b[0] = (__bf16)rb0[j];
        pk.b[1] = (__bf16)rb2[j];
        *(unsigned*)&Bs[nxt][bn0 + j][bkr] = pk.u;
      }
      #pragma unroll
      for (int j = 0; j < 4; ++j) {
        PackU pk;
        pk.b[0] = (__bf16)rb1[j];
        pk.b[1] = (__bf16)rb3[j];
        *(unsigned*)&Bs[nxt][bn0 + 4 + j][bkr] = pk.u;
      }
    }
    __syncthreads();
    p = nxt;
  }

  // --- epilogue (C layout: VGPR r -> M=r(+8 upper half), lane -> N) ---
  #pragma unroll
  for (int mi = 0; mi < 2; ++mi)
    #pragma unroll
    for (int ni = 0; ni < 4; ++ni) {
      const int col = bn + wn + ni * 16 + ln;
      #pragma unroll
      for (int r = 0; r < 8; ++r) {
        const int rowm = bm + wm + mi * 16 + r + ((lane >= 16) ? 8 : 0);
        C[(size_t)rowm * N + col] = acc[mi][ni][r];
      }
    }
}

// ---------------------------------------------------------------------------
// RoPE on Q (fp32 [S, NH, HD] -> bf16 [NH, S, HD])
// ---------------------------------------------------------------------------
__global__ __launch_bounds__(256) void rope_q_kernel(
    const float* __restrict__ q, const int* __restrict__ pos,
    __bf16* __restrict__ qh) {
  int idx = blockIdx.x * 256 + threadIdx.x;       // S*NH*HD elements
  int d = idx & 255;
  int h = (idx >> 8) & 15;
  int s = idx >> 12;
  int dh = d & 127;
  float inv = __expf(-(float)(2 * dh) * (9.210340372f / 256.f)); // 10000^(-2dh/256)
  float ang = (float)pos[s] * inv;
  float c = __cosf(ang), sn = __sinf(ang);
  float val = q[idx];
  float other = (d < 128) ? -q[idx + 128] : q[idx - 128];
  qh[((size_t)h * cS + s) * cHD + d] = (__bf16)(val * c + other * sn);
}

// ---------------------------------------------------------------------------
// RoPE on K + cast V (fp32 [S, NKV, HD] -> bf16 [NKV, S, HD])
// ---------------------------------------------------------------------------
__global__ __launch_bounds__(256) void rope_kv_kernel(
    const float* __restrict__ k, const float* __restrict__ v,
    const int* __restrict__ pos, __bf16* __restrict__ kh,
    __bf16* __restrict__ vh) {
  int idx = blockIdx.x * 256 + threadIdx.x;       // S*NKV*HD elements
  int d = idx & 255;
  int hkv = (idx >> 8) & 7;
  int s = idx >> 11;
  int dh = d & 127;
  float inv = __expf(-(float)(2 * dh) * (9.210340372f / 256.f));
  float ang = (float)pos[s] * inv;
  float c = __cosf(ang), sn = __sinf(ang);
  float kv = k[idx];
  float other = (d < 128) ? -k[idx + 128] : k[idx - 128];
  size_t dst = ((size_t)hkv * cS + s) * cHD + d;
  kh[dst] = (__bf16)(kv * c + other * sn);
  vh[dst] = (__bf16)v[idx];
}

// ---------------------------------------------------------------------------
// Flash-style attention with logit softcap + causal sliding-window mask.
// Grid (S/64, NH), block 128 (4 waves); each wave owns 16 query rows.
// QK^T and PV both via v_wmma_f32_16x16x32_bf16.
// ---------------------------------------------------------------------------
__global__ __launch_bounds__(128) void attn_kernel(
    const __bf16* __restrict__ qh,   // [NH][S][HD]
    const __bf16* __restrict__ kh,   // [NKV][S][HD]
    const __bf16* __restrict__ vh,   // [NKV][S][HD]
    __bf16* __restrict__ attn) {     // [S][NH*HD]
  __shared__ __align__(32) __bf16 Ksm[64][cHD];     // 32 KB
  __shared__ __align__(32) __bf16 Vsm[cHD][64];     // 32 KB (transposed V)
  __shared__ __align__(32) __bf16 Psm[4][16][64];   // 8 KB  (per-wave P)

  const int qtile = blockIdx.x;
  const int h     = blockIdx.y;
  const int kv    = h >> 1;                          // GQA rep = 2
  const int tid   = threadIdx.x;
  const int wv    = tid >> 5;
  const int lane  = tid & 31;
  const int ln    = lane & 15;
  const int hf    = lane >> 4;
  const int qbase = qtile * 64 + wv * 16;
  const int kbA   = hf ? 16 : 0;
  const int kbB   = hf ? 16 : 0;

  // --- preload this wave's Q tile (16x256) as 8 A-fragments ---
  v16bf qf[8];
  const __bf16* qrow = qh + ((size_t)h * cS + qbase + ln) * cHD;
  #pragma unroll
  for (int ks = 0; ks < 8; ++ks) {
    FragU u;
    u.p.lo = *(const v8bf*)(qrow + ks * 32 + kbA);
    u.p.hi = *(const v8bf*)(qrow + ks * 32 + kbA + 8);
    qf[ks] = u.v;
  }

  v8f o[16] = {};
  float mstate[8], lstate[8];
  #pragma unroll
  for (int r = 0; r < 8; ++r) { mstate[r] = cNEG; lstate[r] = 0.f; }

  const int ntiles = qtile + 1;                       // causal tile skip
  for (int kt = 0; kt < ntiles; ++kt) {
    __syncthreads();
    // --- stage K tile 64x256 (b128 both ways) ---
    {
      int row = tid >> 1, c0 = (tid & 1) * 128;
      const __bf16* src = kh + ((size_t)kv * cS + kt * 64 + row) * cHD + c0;
      #pragma unroll
      for (int j = 0; j < 8; ++j)
        *(f32x4*)&Ksm[row][c0 + j * 16] = *(const f32x4*)(src + j * 16);
    }
    // --- stage V tile transposed: pack (kk, kk+1) pairs -> ds_store_b32 ---
    {
      #pragma unroll
      for (int rsp = 0; rsp < 8; ++rsp) {
        int c = rsp * 128 + tid;                      // pair-task id
        int kk = (c >> 5) * 2;                        // even key row
        int d0 = (c & 31) * 8;                        // 8 head dims
        const __bf16* s0 = vh + ((size_t)kv * cS + kt * 64 + kk) * cHD + d0;
        v8bf r0 = *(const v8bf*)(s0);
        v8bf r1 = *(const v8bf*)(s0 + cHD);
        #pragma unroll
        for (int j = 0; j < 8; ++j) {
          PackU pk;
          pk.b[0] = r0[j];
          pk.b[1] = r1[j];
          *(unsigned*)&Vsm[d0 + j][kk] = pk.u;
        }
      }
    }
    __syncthreads();

    // --- scores: S = Q (16x256) x K^T -> 16x64 ---
    v8f sacc[4] = {};
    #pragma unroll
    for (int nt = 0; nt < 4; ++nt) {
      #pragma unroll
      for (int ks = 0; ks < 8; ++ks) {
        v16bf b = *(const v16bf*)&Ksm[nt * 16 + ln][ks * 32 + kbB];
        sacc[nt] = __builtin_amdgcn_wmma_f32_16x16x32_bf16(
            false, qf[ks], false, b, (short)0, sacc[nt], false, false);
      }
    }
    // --- scale + softcap + causal/sliding-window mask ---
    #pragma unroll
    for (int nt = 0; nt < 4; ++nt) {
      const int j = kt * 64 + nt * 16 + ln;
      #pragma unroll
      for (int r = 0; r < 8; ++r) {
        const int i = qbase + r + hf * 8;
        float sv = sacc[nt][r] * cSCALE;
        sv = tanhf(sv * (1.f / cSOFTCAP)) * cSOFTCAP;
        bool ok = (j <= i) && (j > i - cWINDOW);
        sacc[nt][r] = ok ? sv : cNEG;
      }
    }
    // --- online softmax (rows split across 16-lane halves per C layout) ---
    #pragma unroll
    for (int r = 0; r < 8; ++r) {
      float rm = fmaxf(fmaxf(sacc[0][r], sacc[1][r]),
                       fmaxf(sacc[2][r], sacc[3][r]));
      rm = fmaxf(rm, __shfl_xor(rm, 1, 32));
      rm = fmaxf(rm, __shfl_xor(rm, 2, 32));
      rm = fmaxf(rm, __shfl_xor(rm, 4, 32));
      rm = fmaxf(rm, __shfl_xor(rm, 8, 32));
      float mnew = fmaxf(mstate[r], rm);
      float corr = __expf(mstate[r] - mnew);
      mstate[r] = mnew;
      float rs = 0.f;
      #pragma unroll
      for (int nt = 0; nt < 4; ++nt) {
        float p = __expf(sacc[nt][r] - mnew);
        sacc[nt][r] = p;
        rs += p;
      }
      rs += __shfl_xor(rs, 1, 32);
      rs += __shfl_xor(rs, 2, 32);
      rs += __shfl_xor(rs, 4, 32);
      rs += __shfl_xor(rs, 8, 32);
      lstate[r] = lstate[r] * corr + rs;
      #pragma unroll
      for (int ont = 0; ont < 16; ++ont) o[ont][r] *= corr;
    }
    // --- reshape P: C layout -> A layout via per-wave LDS scratch ---
    #pragma unroll
    for (int nt = 0; nt < 4; ++nt)
      #pragma unroll
      for (int r = 0; r < 8; ++r)
        Psm[wv][r + hf * 8][nt * 16 + ln] = (__bf16)sacc[nt][r];
    // DS ops are in-order within a wave; compiler inserts s_wait_dscnt.

    // --- O += P (16x64) x V (64x256) ---
    #pragma unroll
    for (int ks2 = 0; ks2 < 2; ++ks2) {
      FragU ua;
      ua.p.lo = *(const v8bf*)&Psm[wv][ln][ks2 * 32 + kbA];
      ua.p.hi = *(const v8bf*)&Psm[wv][ln][ks2 * 32 + kbA + 8];
      #pragma unroll
      for (int ont = 0; ont < 16; ++ont) {
        v16bf b = *(const v16bf*)&Vsm[ont * 16 + ln][ks2 * 32 + kbB];
        o[ont] = __builtin_amdgcn_wmma_f32_16x16x32_bf16(
            false, ua.v, false, b, (short)0, o[ont], false, false);
      }
    }
  }

  // --- normalize and store attn output (bf16, [S][NH*HD]) ---
  #pragma unroll
  for (int r = 0; r < 8; ++r) {
    float inv = 1.f / lstate[r];
    int row = qbase + r + hf * 8;
    __bf16* dst = attn + (size_t)row * cDQ + h * cHD;
    #pragma unroll
    for (int ont = 0; ont < 16; ++ont)
      dst[ont * 16 + ln] = (__bf16)(o[ont][r] * inv);
  }
}

// ---------------------------------------------------------------------------
// gu = gelu_tanh(G) * U  -> bf16 (down-proj activation)
// ---------------------------------------------------------------------------
__global__ __launch_bounds__(256) void gelu_mul_kernel(
    const float* __restrict__ G, const float* __restrict__ U,
    __bf16* __restrict__ out, long n) {
  long i = (long)blockIdx.x * 256 + threadIdx.x;
  long stride = (long)gridDim.x * 256;
  for (; i < n; i += stride) {
    float g = G[i];
    float u = U[i];
    float t = tanhf(0.7978845608f * (g + 0.044715f * g * g * g));
    out[i] = (__bf16)(0.5f * g * (1.f + t) * u);
  }
}

// ---------------------------------------------------------------------------
// launcher
// ---------------------------------------------------------------------------
extern "C" void kernel_launch(void* const* d_in, const int* in_sizes, int n_in,
                              void* d_out, int out_size, void* d_ws, size_t ws_size,
                              hipStream_t stream) {
  (void)in_sizes; (void)n_in; (void)out_size; (void)ws_size;
  const float* x          = (const float*)d_in[0];
  const int*   pos        = (const int*)d_in[1];
  /* d_in[2] attention_mask: recomputed analytically (causal & window) */
  const float* w_in       = (const float*)d_in[3];
  const float* w_post_at  = (const float*)d_in[4];
  const float* w_pre_ff   = (const float*)d_in[5];
  const float* w_post_ff  = (const float*)d_in[6];
  const float* wq         = (const float*)d_in[7];
  const float* wk         = (const float*)d_in[8];
  const float* wv         = (const float*)d_in[9];
  const float* wo         = (const float*)d_in[10];
  const float* w_gate     = (const float*)d_in[11];
  const float* w_up       = (const float*)d_in[12];
  const float* w_down     = (const float*)d_in[13];
  float* out = (float*)d_out;
  char*  ws  = (char*)d_ws;

  // workspace layout (all sizes 256B-multiples); G aliases the attention
  // temporaries (exactly S*I*4 bytes), dn aliases G after the gelu pass.
  const size_t o_h    = 0;                                   // bf16 S*H
  const size_t o_q    = o_h    + (size_t)cS * cH   * 2;      // f32  S*4096
  const size_t o_k    = o_q    + (size_t)cS * cDQ  * 4;      // f32  S*2048
  const size_t o_v    = o_k    + (size_t)cS * cDKV * 4;      // f32  S*2048
  const size_t o_qh   = o_v    + (size_t)cS * cDKV * 4;      // bf16 S*4096
  const size_t o_kh   = o_qh   + (size_t)cS * cDQ  * 2;      // bf16 S*2048
  const size_t o_vh   = o_kh   + (size_t)cS * cDKV * 2;      // bf16 S*2048
  const size_t o_attn = o_vh   + (size_t)cS * cDKV * 2;      // bf16 S*4096
  const size_t o_ap   = o_attn + (size_t)cS * cDQ  * 2;      // f32  S*H
  const size_t o_x1   = o_ap   + (size_t)cS * cH   * 4;      // f32  S*H
  const size_t o_U    = o_x1   + (size_t)cS * cH   * 4;      // f32  S*I
  const size_t o_gu   = o_U    + (size_t)cS * cI   * 4;      // bf16 S*I
  const size_t o_G    = o_q;   // region [o_q, o_ap) == S*I*4 bytes exactly
  const size_t o_dn   = o_q;   // reused again after gelu consumes G

  __bf16* h_bf   = (__bf16*)(ws + o_h);
  float*  q_f    = (float*)(ws + o_q);
  float*  k_f    = (float*)(ws + o_k);
  float*  v_f    = (float*)(ws + o_v);
  __bf16* qh     = (__bf16*)(ws + o_qh);
  __bf16* kh     = (__bf16*)(ws + o_kh);
  __bf16* vh     = (__bf16*)(ws + o_vh);
  __bf16* attnb  = (__bf16*)(ws + o_attn);
  float*  aproj  = (float*)(ws + o_ap);
  float*  x1     = (float*)(ws + o_x1);
  float*  Gf     = (float*)(ws + o_G);
  float*  Uf     = (float*)(ws + o_U);
  __bf16* gu     = (__bf16*)(ws + o_gu);
  float*  dn     = (float*)(ws + o_dn);

  // ---- attention block ----
  rmsnorm_bf16_kernel<<<cS, 256, 0, stream>>>(x, w_in, h_bf, cH);
  wmma_gemm_kernel<<<dim3(cDQ / 128, cS / 128), 256, 0, stream>>>(h_bf, wq, q_f, cS, cDQ, cH);
  wmma_gemm_kernel<<<dim3(cDKV / 128, cS / 128), 256, 0, stream>>>(h_bf, wk, k_f, cS, cDKV, cH);
  wmma_gemm_kernel<<<dim3(cDKV / 128, cS / 128), 256, 0, stream>>>(h_bf, wv, v_f, cS, cDKV, cH);
  rope_q_kernel<<<(cS * cDQ) / 256, 256, 0, stream>>>(q_f, pos, qh);
  rope_kv_kernel<<<(cS * cDKV) / 256, 256, 0, stream>>>(k_f, v_f, pos, kh, vh);
  attn_kernel<<<dim3(cS / 64, cNH), 128, 0, stream>>>(qh, kh, vh, attnb);
  wmma_gemm_kernel<<<dim3(cH / 128, cS / 128), 256, 0, stream>>>(attnb, wo, aproj, cS, cH, cDQ);
  add_rmsnorm_kernel<<<cS, 256, 0, stream>>>(x, aproj, w_post_at, x1, cH);

  // ---- MLP block ----
  rmsnorm_bf16_kernel<<<cS, 256, 0, stream>>>(x1, w_pre_ff, h_bf, cH);
  wmma_gemm_kernel<<<dim3(cI / 128, cS / 128), 256, 0, stream>>>(h_bf, w_gate, Gf, cS, cI, cH);
  wmma_gemm_kernel<<<dim3(cI / 128, cS / 128), 256, 0, stream>>>(h_bf, w_up, Uf, cS, cI, cH);
  gelu_mul_kernel<<<8192, 256, 0, stream>>>(Gf, Uf, gu, (long)cS * cI);
  wmma_gemm_kernel<<<dim3(cH / 128, cS / 128), 256, 0, stream>>>(gu, w_down, dn, cS, cH, cI);
  add_rmsnorm_kernel<<<cS, 256, 0, stream>>>(x1, dn, w_post_ff, out, cH);
}
